// GenericLSTM_81844896793006
// MI455X (gfx1250) — matrix operand order repaired and checked by reading, not compiled, not run on hardware
//
#include <hip/hip_runtime.h>
#include <hip/hip_bf16.h>

// ---------------------------------------------------------------------------
// Persistent-LSTM for MI455X (gfx1250, wave32, WMMA).
//   B=64, T=2048, I=128, H=256, 4 gates.
//   Pre-pass: x f32 -> bf16 (streaming, parallel) into d_ws.
//   8 persistent workgroups, each owns 32 h-columns (x 4 gates).
//   Weights (bf16) resident in LDS (96 KB/WGP), cell state resident in VGPRs.
//   Recurrent h broadcast via double-buffered bf16 buffer in d_ws + a
//   device-scope arrival counter per step.
// ---------------------------------------------------------------------------

typedef __attribute__((ext_vector_type(16))) __bf16 v16bf;
typedef __attribute__((ext_vector_type(8)))  float  v8f;

constexpr int Bb   = 64;
constexpr int Tt   = 2048;
constexpr int Ii   = 128;
constexpr int Hh   = 256;
constexpr int Gg   = 4;
constexpr int NWG  = 8;          // persistent workgroups (one per WGP)
constexpr int COLS = Hh / NWG;   // 32 h-columns per WG
constexpr int THREADS = 256;     // 8 waves

union ABFrag {
    v16bf v;
    uint4 q[2];
    unsigned u[8];
};

__device__ __forceinline__ unsigned f32_to_bf16u(float a) {
    unsigned u = __float_as_uint(a);
    u += 0x7FFFu + ((u >> 16) & 1u);      // round-to-nearest-even
    return u >> 16;
}
__device__ __forceinline__ unsigned pack_bf16x2(float a, float b) {
    unsigned ua = __float_as_uint(a); ua += 0x7FFFu + ((ua >> 16) & 1u);
    unsigned ub = __float_as_uint(b); ub += 0x7FFFu + ((ub >> 16) & 1u);
    return (ua >> 16) | (ub & 0xFFFF0000u);
}
__device__ __forceinline__ float fast_sigmoid(float x) {
    return __frcp_rn(1.0f + __expf(-x));
}
__device__ __forceinline__ float fast_tanh(float x) {
    float e = __expf(-2.0f * x);
    return (1.0f - e) * __frcp_rn(1.0f + e);
}

// Zero the h double-buffer and the sync counter each launch (replay-safe).
__global__ void lstm_init_kernel(unsigned* ws, int n) {
    int i = blockIdx.x * blockDim.x + threadIdx.x;
    if (i < n) ws[i] = 0u;
}

// Streaming f32 -> bf16 conversion of x (8 elements / thread, b128 in/out).
__global__ __launch_bounds__(256)
void x_to_bf16_kernel(const float* __restrict__ x,
                      unsigned short* __restrict__ xbf, int n8) {
    int i = blockIdx.x * blockDim.x + threadIdx.x;
    if (i < n8) {
        const float4* p = (const float4*)x + (size_t)2 * i;
        float4 f0 = p[0];
        float4 f1 = p[1];
        uint4 o;
        o.x = pack_bf16x2(f0.x, f0.y);
        o.y = pack_bf16x2(f0.z, f0.w);
        o.z = pack_bf16x2(f1.x, f1.y);
        o.w = pack_bf16x2(f1.z, f1.w);
        ((uint4*)xbf)[i] = o;
    }
}

template <bool XPRE>   // XPRE: x already converted to bf16 in d_ws
__global__ __launch_bounds__(THREADS)
void lstm_persistent_kernel(const float* __restrict__ x,            // [B,T,I] f32
                            const unsigned short* __restrict__ xbf, // [B,T,I] bf16
                            const float* __restrict__ Wx,           // [4,I,H]
                            const float* __restrict__ Wh,           // [4,H,H]
                            const float* __restrict__ bias,         // [4,H]
                            float* __restrict__ out,                // [B,T,H]
                            unsigned short* __restrict__ hbuf,      // [2][B][H] bf16
                            unsigned* __restrict__ syncCnt) {
    extern __shared__ char smem[];
    unsigned short* sWh = (unsigned short*)smem;            // [4][COLS][H]  64 KB
    unsigned short* sWx = sWh + Gg * COLS * Hh;             // [4][COLS][I]  32 KB

    const int wg   = blockIdx.x;       // 0..7 -> h-columns [wg*32, wg*32+32)
    const int tid  = threadIdx.x;
    const int lane = tid & 31;
    const int wave = tid >> 5;         // 0..7
    const int bt   = wave & 3;         // B-tile (rows 16*bt..)
    const int hh   = wave >> 2;        // which 16-column half of the WG's 32
    const int mrow = lane & 15;        // A-matrix row within tile (both halves)
    const int hi   = lane >> 4;        // lane-half selector for K chunks
    const int col  = wg * COLS + hh * 16 + mrow;   // this lane's global h-column
    const int laneColOff = hh * 16 + mrow;         // column within WG slice

    // ---- Stage weight slices into LDS as bf16, column-major in K ----------
    for (int idx = tid; idx < Gg * Hh * COLS; idx += THREADS) {
        int g = idx / (Hh * COLS);
        int r = idx % (Hh * COLS);
        int k = r / COLS;
        int c = r % COLS;
        float v = Wh[(size_t)g * Hh * Hh + (size_t)k * Hh + wg * COLS + c];
        sWh[g * (COLS * Hh) + c * Hh + k] = (unsigned short)f32_to_bf16u(v);
    }
    for (int idx = tid; idx < Gg * Ii * COLS; idx += THREADS) {
        int g = idx / (Ii * COLS);
        int r = idx % (Ii * COLS);
        int k = r / COLS;
        int c = r % COLS;
        float v = Wx[(size_t)g * Ii * Hh + (size_t)k * Hh + wg * COLS + c];
        sWx[g * (COLS * Ii) + c * Ii + k] = (unsigned short)f32_to_bf16u(v);
    }
    __syncthreads();

    // ---- Per-lane bias (C/D layout: col = lane&15, constant over VGPRs) ---
    float bias_g[Gg];
#pragma unroll
    for (int g = 0; g < Gg; ++g) bias_g[g] = bias[g * Hh + col];

    v8f cstate;
#pragma unroll
    for (int e = 0; e < 8; ++e) cstate[e] = 0.0f;

    const int arow = bt * 16 + mrow;   // global batch row for A fragments

    for (int t = 0; t < Tt; ++t) {
        const unsigned short* hprev = hbuf + (size_t)(t & 1) * Bb * Hh;
        unsigned short* hnext = hbuf + (size_t)((t + 1) & 1) * Bb * Hh;

        v8f acc[Gg];
#pragma unroll
        for (int g = 0; g < Gg; ++g)
#pragma unroll
            for (int e = 0; e < 8; ++e) acc[g][e] = bias_g[g];

        // ---- x_t projection: K = 128 (4 WMMA K-tiles) ---------------------
#pragma unroll
        for (int kt = 0; kt < Ii / 32; ++kt) {
            ABFrag a;
            const int k0 = kt * 32 + hi * 8;   // ISA 16-bit A layout chunks
            if constexpr (XPRE) {
                const unsigned short* xp =
                    xbf + (size_t)arow * Tt * Ii + (size_t)t * Ii;
                if (kt == 0) __builtin_prefetch(xp + Ii, 0, 0);  // next step
                a.q[0] = *(const uint4*)(xp + k0);        // K   .. K+7
                a.q[1] = *(const uint4*)(xp + k0 + 16);   // K+16.. K+23
            } else {
                const float* xp = x + (size_t)arow * Tt * Ii + (size_t)t * Ii;
                if (kt == 0) __builtin_prefetch(xp + Ii, 0, 0);
                float4 f0 = *(const float4*)(xp + k0);
                float4 f1 = *(const float4*)(xp + k0 + 4);
                float4 f2 = *(const float4*)(xp + k0 + 16);
                float4 f3 = *(const float4*)(xp + k0 + 20);
                a.u[0] = pack_bf16x2(f0.x, f0.y);
                a.u[1] = pack_bf16x2(f0.z, f0.w);
                a.u[2] = pack_bf16x2(f1.x, f1.y);
                a.u[3] = pack_bf16x2(f1.z, f1.w);
                a.u[4] = pack_bf16x2(f2.x, f2.y);
                a.u[5] = pack_bf16x2(f2.z, f2.w);
                a.u[6] = pack_bf16x2(f3.x, f3.y);
                a.u[7] = pack_bf16x2(f3.z, f3.w);
            }
#pragma unroll
            for (int g = 0; g < Gg; ++g) {
                ABFrag b;
                const uint4* bp = (const uint4*)(sWx + g * (COLS * Ii) +
                                                 laneColOff * Ii + kt * 32 + hi * 8);
                b.q[0] = bp[0];
                b.q[1] = bp[2];
                acc[g] = __builtin_amdgcn_wmma_f32_16x16x32_bf16(
                    false, a.v, false, b.v, (short)0, acc[g], false, false);
            }
        }

        // ---- h_{t-1} projection: K = 256 (8 WMMA K-tiles), bf16 in hbuf ---
        const unsigned short* hp = hprev + (size_t)arow * Hh;
#pragma unroll
        for (int kt = 0; kt < Hh / 32; ++kt) {
            ABFrag a;
            const int k0 = kt * 32 + hi * 8;
            a.q[0] = *(const uint4*)(hp + k0);        // K   .. K+7
            a.q[1] = *(const uint4*)(hp + k0 + 16);   // K+16.. K+23
#pragma unroll
            for (int g = 0; g < Gg; ++g) {
                ABFrag b;
                const uint4* bp = (const uint4*)(sWh + g * (COLS * Hh) +
                                                 laneColOff * Hh + kt * 32 + hi * 8);
                b.q[0] = bp[0];
                b.q[1] = bp[2];
                acc[g] = __builtin_amdgcn_wmma_f32_16x16x32_bf16(
                    false, a.v, false, b.v, (short)0, acc[g], false, false);
            }
        }

        // ---- Elementwise LSTM update; c stays in VGPRs --------------------
        float hval[8];
#pragma unroll
        for (int e = 0; e < 8; ++e) {
            float ig = fast_sigmoid(acc[0][e]);
            float fg = fast_sigmoid(acc[1][e]);
            float gg = fast_tanh(acc[2][e]);
            float og = fast_sigmoid(acc[3][e]);
            float c  = fg * cstate[e] + ig * gg;
            cstate[e] = c;
            hval[e] = og * fast_tanh(c);
        }

        // ---- Store h: f32 to out, bf16 to broadcast buffer ----------------
        // C/D layout: VGPR e -> row M = e + 8*(lane>=16), col = lane&15.
#pragma unroll
        for (int e = 0; e < 8; ++e) {
            int brow = bt * 16 + e + hi * 8;
            out[(size_t)brow * Tt * Hh + (size_t)t * Hh + col] = hval[e];
            hnext[(size_t)brow * Hh + col] = (unsigned short)f32_to_bf16u(hval[e]);
        }

        // ---- Device-scope step barrier across the 8 workgroups ------------
        if (t + 1 < Tt) {
            __threadfence();
            __syncthreads();
            if (tid == 0) {
                __hip_atomic_fetch_add(syncCnt, 1u, __ATOMIC_RELEASE,
                                       __HIP_MEMORY_SCOPE_AGENT);
                const unsigned target = (unsigned)(NWG * (t + 1));
                while (__hip_atomic_load(syncCnt, __ATOMIC_ACQUIRE,
                                         __HIP_MEMORY_SCOPE_AGENT) < target) {
                    __builtin_amdgcn_s_sleep(2);
                }
            }
            __syncthreads();
            __threadfence();
        }
    }
}

extern "C" void kernel_launch(void* const* d_in, const int* in_sizes, int n_in,
                              void* d_out, int out_size, void* d_ws, size_t ws_size,
                              hipStream_t stream) {
    const float* x    = (const float*)d_in[0];  // [B,T,I]
    const float* Wx   = (const float*)d_in[1];  // [4,I,H]
    const float* Wh   = (const float*)d_in[2];  // [4,H,H]
    const float* bias = (const float*)d_in[3];  // [4,H]
    float* out = (float*)d_out;                 // [B,T,H]

    const size_t xbfBytes  = (size_t)Bb * Tt * Ii * sizeof(unsigned short); // 32 MB
    const size_t hbufBytes = (size_t)2 * Bb * Hh * sizeof(unsigned short);  // 64 KB
    const size_t needBig   = xbfBytes + hbufBytes + 64;

    const size_t shmem = (size_t)(Gg * COLS * Hh + Gg * COLS * Ii) *
                         sizeof(unsigned short);  // 96 KB

    if (ws_size >= needBig) {
        // Path A: pre-convert x to bf16 (parallel stream), then persistent LSTM.
        unsigned short* xbf  = (unsigned short*)d_ws;
        unsigned short* hbuf = (unsigned short*)((char*)d_ws + xbfBytes);
        unsigned* syncCnt    = (unsigned*)((char*)d_ws + xbfBytes + hbufBytes);

        const int n8 = Bb * Tt * Ii / 8;   // 2,097,152 uint4 stores
        x_to_bf16_kernel<<<(n8 + 255) / 256, 256, 0, stream>>>(x, xbf, n8);

        const int initDwords = (int)((hbufBytes + 64) / 4);
        lstm_init_kernel<<<(initDwords + 255) / 256, 256, 0, stream>>>(
            (unsigned*)hbuf, initDwords);

        lstm_persistent_kernel<true><<<dim3(NWG), dim3(THREADS), shmem, stream>>>(
            x, xbf, Wx, Wh, bias, out, hbuf, syncCnt);
    } else {
        // Path B (small scratch): convert x in-loop inside the persistent kernel.
        unsigned short* hbuf = (unsigned short*)d_ws;
        unsigned* syncCnt    = (unsigned*)((char*)d_ws + hbufBytes);

        const int initDwords = (int)((hbufBytes + 64) / 4);
        lstm_init_kernel<<<(initDwords + 255) / 256, 256, 0, stream>>>(
            (unsigned*)hbuf, initDwords);

        lstm_persistent_kernel<false><<<dim3(NWG), dim3(THREADS), shmem, stream>>>(
            x, (const unsigned short*)nullptr, Wx, Wh, bias, out, hbuf, syncCnt);
    }
}
